// SDE_36206574305446
// MI455X (gfx1250) — matrix operand order repaired and checked by reading, not compiled, not run on hardware
//
#include <hip/hip_runtime.h>
#include <hip/hip_bf16.h>

typedef __attribute__((ext_vector_type(16))) __bf16 v16bf;
typedef __attribute__((ext_vector_type(8)))  float  v8f;

#define W_TOT   16000
#define D_      128
#define L_      10000
#define NSPEC   4
#define T_TOT   384000
#define NTOK    8192        // B*S = 8*1024
#define LB      32          // latent rows per block iteration
#define NIT     ((L_ + LB - 1) / LB)   // 313

static __device__ __forceinline__ unsigned short f2bf(float f) {
    unsigned int u = __float_as_uint(f);
    unsigned int r = u + 0x7FFFu + ((u >> 16) & 1u);   // RNE
    return (unsigned short)(r >> 16);
}
static __device__ __forceinline__ float bf2f(unsigned short b) {
    return __uint_as_float(((unsigned int)b) << 16);
}
static __device__ __forceinline__ float fast_tanh(float x) {
    x = fminf(15.f, fmaxf(-15.f, x));
    float e2 = __expf(2.f * x);
    return (e2 - 1.f) / (e2 + 1.f);
}
static __device__ __forceinline__ v16bf mk16(uint4 lo, uint4 hi) {
    union { unsigned int u[8]; v16bf v; } x;
    x.u[0] = lo.x; x.u[1] = lo.y; x.u[2] = lo.z; x.u[3] = lo.w;
    x.u[4] = hi.x; x.u[5] = hi.y; x.u[6] = hi.z; x.u[7] = hi.w;
    return x.v;
}

// ---------------------------------------------------------------------------
// Kernel B: EmbeddingBag(sum). One wave per 32 sorted rows; lane owns a
// float4 d-slice. Run-length accumulate, flush with global f32 atomics.
// ---------------------------------------------------------------------------
__global__ __launch_bounds__(256) void bag_kernel(
        const int* __restrict__ ngram_ids, const int* __restrict__ segment_ids,
        const float* __restrict__ emb, float* __restrict__ bag) {
    int w = (blockIdx.x * blockDim.x + threadIdx.x) >> 5;
    int lane = threadIdx.x & 31;
    if (w >= T_TOT / 32) return;
    int t0 = w * 32;
    int d  = lane * 4;
    float4 a = make_float4(0.f, 0.f, 0.f, 0.f);
    int cur = segment_ids[t0];
    #pragma unroll 4
    for (int i = 0; i < 32; ++i) {
        int t   = t0 + i;
        int seg = segment_ids[t];
        int row = ngram_ids[t];
        float4 v = *(const float4*)(emb + (size_t)row * D_ + d);
        if (seg != cur) {
            float* b = bag + (size_t)cur * D_ + d;
            atomicAdd(b + 0, a.x); atomicAdd(b + 1, a.y);
            atomicAdd(b + 2, a.z); atomicAdd(b + 3, a.w);
            a = make_float4(0.f, 0.f, 0.f, 0.f);
            cur = seg;
        }
        a.x += v.x; a.y += v.y; a.z += v.z; a.w += v.w;
    }
    float* b = bag + (size_t)cur * D_ + d;
    atomicAdd(b + 0, a.x); atomicAdd(b + 1, a.y);
    atomicAdd(b + 2, a.z); atomicAdd(b + 3, a.w);
}

// ---------------------------------------------------------------------------
// Kernel C: fused tanh + flash-softmax codebook lookup, bf16 WMMA, f32 acc.
// Block = 128 threads = 4 waves; wave owns a 16-row word tile.
// Latent block kept in LDS in two layouts (row-major + d-major) so every
// WMMA fragment is a pair of contiguous uint4 LDS loads.
// ---------------------------------------------------------------------------
__global__ __launch_bounds__(128) void sde_latent_kernel(
        const float* __restrict__ bag, const float* __restrict__ latent,
        float* __restrict__ token) {
    __shared__ __align__(16) unsigned short latB [LB][D_];   // [l][d]
    __shared__ __align__(16) unsigned short latBT[D_][LB];   // [d][l]
    __shared__ __align__(16) unsigned short nwS[4][16][D_];
    __shared__ __align__(16) float          scoreS[4][16][LB];
    __shared__ __align__(16) unsigned short pS[4][16][LB];
    __shared__ float scaleS[4][16];
    __shared__ float sumS[4][16];

    const int wv   = threadIdx.x >> 5;
    const int lane = threadIdx.x & 31;
    const int wt   = blockIdx.x * 4 + wv;          // word tile id (0..999)
    const int mrow = lane & 15;
    const int hi16 = lane >> 4;                    // 0 or 1 (lane half)

    // --- load nw = tanh(bag) into LDS as bf16 -----------------------------
    {
        int r  = lane >> 1;            // 0..15
        int d0 = (lane & 1) * 64;
        const float4* src = (const float4*)(bag + ((size_t)(wt * 16 + r)) * D_ + d0);
        uint2* dst = (uint2*)(&nwS[wv][r][d0]);
        #pragma unroll
        for (int i = 0; i < 16; ++i) {
            float4 v = src[i];
            v.x = fast_tanh(v.x); v.y = fast_tanh(v.y);
            v.z = fast_tanh(v.z); v.w = fast_tanh(v.w);
            uint2 p;
            p.x = (unsigned)f2bf(v.x) | ((unsigned)f2bf(v.y) << 16);
            p.y = (unsigned)f2bf(v.z) | ((unsigned)f2bf(v.w) << 16);
            dst[i] = p;
        }
    }
    __syncthreads();

    // --- hoist nw A-fragments (loop invariant), 4 K-chunks of 32 ----------
    v16bf nwA[4];
    {
        int kb = hi16 * 8;             // A layout: K-runs of 8 per lane half
        #pragma unroll
        for (int kc = 0; kc < 4; ++kc) {
            int base = kc * 32 + kb;
            uint4 lo = *(const uint4*)(&nwS[wv][mrow][base]);
            uint4 h  = *(const uint4*)(&nwS[wv][mrow][base + 16]);
            nwA[kc] = mk16(lo, h);
        }
    }

    v8f acc[8];
    #pragma unroll
    for (int t = 0; t < 8; ++t) acc[t] = (v8f){0.f,0.f,0.f,0.f,0.f,0.f,0.f,0.f};

    float m_run = -3.0e38f;   // owned by lanes 0..15 (row = lane)
    float s_run = 0.f;

    for (int it = 0; it < NIT; ++it) {
        const int l0 = it * LB;

        // --- cooperative load of latent block into both LDS layouts -------
        {
            int r  = threadIdx.x >> 2;          // 0..31 latent row in block
            int d0 = (threadIdx.x & 3) * 32;
            int gl = l0 + r;
            const float4* src = (const float4*)(latent + (size_t)gl * D_ + d0);
            #pragma unroll
            for (int i = 0; i < 8; ++i) {
                float4 v = (gl < L_) ? src[i] : make_float4(0.f, 0.f, 0.f, 0.f);
                unsigned short b0 = f2bf(v.x), b1 = f2bf(v.y);
                unsigned short b2 = f2bf(v.z), b3 = f2bf(v.w);
                int d = d0 + i * 4;
                uint2 p;
                p.x = (unsigned)b0 | ((unsigned)b1 << 16);
                p.y = (unsigned)b2 | ((unsigned)b3 << 16);
                *(uint2*)(&latB[r][d]) = p;
                latBT[d + 0][r] = b0; latBT[d + 1][r] = b1;
                latBT[d + 2][r] = b2; latBT[d + 3][r] = b3;
            }
        }
        __syncthreads();

        // --- scores[16x32] = nw @ latent_blockᵀ  (2 N-tiles, K=128) -------
        {
            #pragma unroll
            for (int j = 0; j < 2; ++j) {
                v8f c = (v8f){0.f,0.f,0.f,0.f,0.f,0.f,0.f,0.f};
                int lrow = j * 16 + mrow;               // latent row = tile col
                #pragma unroll
                for (int kc = 0; kc < 4; ++kc) {
                    // B layout: K-runs of 8 per lane half (same as A)
                    int base = kc * 32 + hi16 * 8;
                    uint4 blo = *(const uint4*)(&latB[lrow][base]);
                    uint4 bhi = *(const uint4*)(&latB[lrow][base + 16]);
                    v16bf b = mk16(blo, bhi);
                    c = __builtin_amdgcn_wmma_f32_16x16x32_bf16(
                            false, nwA[kc], false, b, (short)0, c, false, false);
                }
                int mb = hi16 * 8;
                #pragma unroll
                for (int i = 0; i < 8; ++i)
                    scoreS[wv][mb + i][j * 16 + mrow] = c[i];
            }
        }
        __syncthreads();

        // --- online softmax: lanes 0..15 own one row each -----------------
        if (lane < 16) {
            int row = lane;
            float sc[LB];
            float bm = -3.0e38f;
            #pragma unroll
            for (int c = 0; c < LB; ++c) {
                float v = scoreS[wv][row][c];
                if (l0 + c >= L_) v = -3.0e38f;
                sc[c] = v;
                bm = fmaxf(bm, v);
            }
            float m_new = fmaxf(m_run, bm);
            float scale = __expf(m_run - m_new);   // 0 on first iteration
            float ssum  = 0.f;
            #pragma unroll
            for (int c = 0; c < LB; ++c) {
                float e = (l0 + c >= L_) ? 0.f : __expf(sc[c] - m_new);
                ssum += e;
                pS[wv][row][c] = f2bf(e);
            }
            s_run = s_run * scale + ssum;
            m_run = m_new;
            scaleS[wv][row] = scale;
        }
        __syncthreads();

        // --- rescale acc and accumulate P @ latent_block ------------------
        {
            float scv[8];
            int mb = hi16 * 8;
            #pragma unroll
            for (int i = 0; i < 8; ++i) scv[i] = scaleS[wv][mb + i];
            #pragma unroll
            for (int t = 0; t < 8; ++t)
                #pragma unroll
                for (int i = 0; i < 8; ++i) acc[t][i] *= scv[i];

            // A fragment from P[16x32] (K-runs of 8 per lane half)
            int kb = hi16 * 8;
            uint4 alo = *(const uint4*)(&pS[wv][mrow][kb]);
            uint4 ahi = *(const uint4*)(&pS[wv][mrow][kb + 16]);
            v16bf a = mk16(alo, ahi);

            int kb2 = hi16 * 8;      // B fragment: K-runs of 8 per lane half
            #pragma unroll
            for (int t = 0; t < 8; ++t) {
                int n = t * 16 + mrow;              // d column
                uint4 blo = *(const uint4*)(&latBT[n][kb2]);
                uint4 bhi = *(const uint4*)(&latBT[n][kb2 + 16]);
                v16bf b = mk16(blo, bhi);
                acc[t] = __builtin_amdgcn_wmma_f32_16x16x32_bf16(
                             false, a, false, b, (short)0, acc[t], false, false);
            }
        }
        __syncthreads();   // protect latB/latBT before next overwrite
    }

    if (lane < 16) sumS[wv][lane] = s_run;
    __syncthreads();

    // --- token_emb = nw + acc / sum ---------------------------------------
    {
        int mb = hi16 * 8;
        float inv[8];
        #pragma unroll
        for (int i = 0; i < 8; ++i) inv[i] = 1.0f / sumS[wv][mb + i];
        #pragma unroll
        for (int t = 0; t < 8; ++t) {
            #pragma unroll
            for (int i = 0; i < 8; ++i) {
                int m = mb + i;
                int d = t * 16 + mrow;
                float nwv = bf2f(nwS[wv][m][d]);
                token[(size_t)(wt * 16 + m) * D_ + d] = nwv + acc[t][i] * inv[i];
            }
        }
    }
}

// ---------------------------------------------------------------------------
// Kernel D: final embedding gather.  out[tok] = concat(special, token)[x[tok]]
// ---------------------------------------------------------------------------
__global__ __launch_bounds__(256) void gather_kernel(
        const int* __restrict__ x, const float* __restrict__ special,
        const float* __restrict__ token, float* __restrict__ out) {
    int gid = blockIdx.x * blockDim.x + threadIdx.x;
    int tok = gid >> 5;
    int d   = (gid & 31) * 4;
    if (tok >= NTOK) return;
    int id = x[tok];
    const float* src = (id < NSPEC)
        ? (special + (size_t)id * D_ + d)
        : (token + (size_t)(id - NSPEC) * D_ + d);
    *(float4*)(out + (size_t)tok * D_ + d) = *(const float4*)src;
}

extern "C" void kernel_launch(void* const* d_in, const int* in_sizes, int n_in,
                              void* d_out, int out_size, void* d_ws, size_t ws_size,
                              hipStream_t stream) {
    const int*   x           = (const int*)d_in[0];
    const int*   ngram_ids   = (const int*)d_in[1];
    const int*   segment_ids = (const int*)d_in[2];
    const float* ngram_emb   = (const float*)d_in[3];
    const float* latent      = (const float*)d_in[4];
    const float* special     = (const float*)d_in[5];
    (void)in_sizes; (void)n_in; (void)out_size; (void)ws_size;

    float* bag   = (float*)d_ws;                      // [16000,128]
    float* token = bag + (size_t)W_TOT * D_;          // [16000,128]

    hipMemsetAsync(bag, 0, (size_t)W_TOT * D_ * sizeof(float), stream);

    bag_kernel<<<(T_TOT / 32) / 8, 256, 0, stream>>>(ngram_ids, segment_ids,
                                                     ngram_emb, bag);
    sde_latent_kernel<<<(W_TOT / 16) / 4, 128, 0, stream>>>(bag, latent, token);
    gather_kernel<<<(NTOK * 32) / 256, 256, 0, stream>>>(x, special, token,
                                                         (float*)d_out);
}
